// SparseMarketMoE_29506425324174
// MI455X (gfx1250) — matrix operand order repaired and checked by reading, not compiled, not run on hardware
//
#include <hip/hip_runtime.h>
#include <hip/hip_bf16.h>
#include <math.h>

typedef __bf16 bf16;
typedef __attribute__((ext_vector_type(16))) __bf16 v16bf;
typedef __attribute__((ext_vector_type(8)))  float  v8f;

#define WMMA_BF16(a,b,c) \
  __builtin_amdgcn_wmma_f32_16x16x32_bf16(false,(a),false,(b),(short)0,(c),false,false)

#define NTOK 65536
#define NE   9
// ws layout in floats
#define WS_CB    0      // router const bias (9, padded 16)
#define WS_SB    16     // style_bias E*128 = 1152
#define WS_PART  1168   // market partials: 256 blocks * 32
#define WS_AUXP  9360   // aux partials: 512 blocks * 18

// ---------------- fragment loaders (ISA 7.12.2 layouts) ----------------
__device__ __forceinline__ v16bf frag_a(const bf16* tile, int lane, int kc) {
  const int hi = lane >> 4, m = lane & 15;
  const int k0 = kc * 32 + hi * 8;            // halves 0..7 -> k0..k0+7, 8..15 -> k0+16..k0+23
  union { v16bf v; uint4 u[2]; } r;
  r.u[0] = *(const uint4*)(tile + m * 128 + k0);
  r.u[1] = *(const uint4*)(tile + m * 128 + k0 + 16);
  return r.v;
}
__device__ __forceinline__ v16bf frag_b(const bf16* wt, int lane, int n0, int kc) {
  const int hi = lane >> 4, n = n0 * 16 + (lane & 15);
  const int k0 = kc * 32 + hi * 8;            // halves 0..7 -> k0..k0+7, 8..15 -> k0+16..k0+23
  union { v16bf v; uint4 u[2]; } r;
  r.u[0] = *(const uint4*)(wt + n * 128 + k0);
  r.u[1] = *(const uint4*)(wt + n * 128 + k0 + 16);
  return r.v;
}

// ---------------- K1: market = relu(x@Wr+bias), weighted partial sums ----------------
__global__ __launch_bounds__(256) void k_market(const float* __restrict__ x,
                                                const float* __restrict__ Wrw,
                                                const float* __restrict__ Wrb,
                                                const float* __restrict__ Wlw,
                                                float* __restrict__ ws) {
  __shared__ float wr[128 * 32];
  __shared__ float part[256][33];
  const int t = threadIdx.x;
  for (int i = t; i < 128 * 32; i += 256) wr[i] = Wrw[i];
  __syncthreads();
  const int n = blockIdx.x * 256 + t;
  float acc[32];
#pragma unroll
  for (int m = 0; m < 32; ++m) acc[m] = 0.f;
  const float4* xr = (const float4*)(x + (size_t)n * 128);
  for (int k4 = 0; k4 < 32; ++k4) {
    float4 xv = xr[k4];
    const float* w0 = &wr[(k4 * 4 + 0) * 32];
#pragma unroll
    for (int m = 0; m < 32; ++m)
      acc[m] += xv.x * w0[m] + xv.y * w0[32 + m] + xv.z * w0[64 + m] + xv.w * w0[96 + m];
  }
  const float wl = Wlw[n];
#pragma unroll
  for (int m = 0; m < 32; ++m) {
    float v = acc[m] + Wrb[m];
    part[t][m] = wl * (v > 0.f ? v : 0.f);
  }
  __syncthreads();
  if (t < 32) {
    float s = 0.f;
    for (int i = 0; i < 256; ++i) s += part[i][t];    // fixed order: deterministic
    ws[WS_PART + blockIdx.x * 32 + t] = s;
  }
}

// ---------------- K2: market_g reduce -> router const bias; style_bias ----------------
__global__ __launch_bounds__(256) void k_setup(const float* __restrict__ Wlb,
                                               const float* __restrict__ router_w,
                                               const float* __restrict__ router_b,
                                               const float* __restrict__ styles,
                                               const float* __restrict__ W1,
                                               const float* __restrict__ b1,
                                               float* __restrict__ ws) {
  __shared__ float mg[32];
  const int t = threadIdx.x;
  if (t < 32) {
    float s = Wlb[0];
    for (int b = 0; b < 256; ++b) s += ws[WS_PART + b * 32 + t];
    mg[t] = s;
  }
  // style_bias[e][h] = sum_s styles[e][s]*W1[e][128+s][h] + b1[e][h]
  for (int i = t; i < NE * 128; i += 256) {
    int e = i >> 7, h = i & 127;
    float acc = b1[i];
    for (int s = 0; s < 32; ++s)
      acc += styles[e * 32 + s] * W1[((size_t)e * 160 + 128 + s) * 128 + h];
    ws[WS_SB + i] = acc;
  }
  __syncthreads();
  if (t < NE) {
    float acc = router_b[t];
    for (int m = 0; m < 32; ++m) acc += mg[m] * router_w[(128 + m) * NE + t];
    ws[WS_CB + t] = acc;
  }
}

// ---------------- K3: gating + dense 9-expert MLP via bf16 WMMA ----------------
__global__ __launch_bounds__(256) void k_moe(const float* __restrict__ x,
                                             const float* __restrict__ router_w,
                                             const float* __restrict__ W1,
                                             const float* __restrict__ W2,
                                             const float* __restrict__ b2,
                                             float* __restrict__ ws,
                                             float* __restrict__ out) {
  __shared__ bf16 w1t[128 * 128];     // W1x[e]^T  [h][k]
  __shared__ bf16 w2t[128 * 128];     // W2[e]^T   [d][k]
  __shared__ bf16 xh[128 * 128];      // x tile (then reused as h, per-wave regions)
  __shared__ bf16 rbt[16 * 128];      // router^T padded to 16 cols
  __shared__ float gat[8][16][12];    // per-wave gating / logits
  __shared__ float sbl[128], b2l[128];
  __shared__ float auxl[18];

  const int t = threadIdx.x, blk = blockIdx.x;
  const int lane = t & 31, w = t >> 5, hi = lane >> 4, lm = lane & 15;

  { // stage x tile as bf16 [tok][k]
    const float* xb = x + (size_t)blk * 128 * 128;
#pragma unroll
    for (int j = 0; j < 16; ++j) {
      int idx = (t + j * 256) * 4;
      float4 v = *(const float4*)(xb + idx);
      bf16* p = &xh[idx];
      p[0] = (bf16)v.x; p[1] = (bf16)v.y; p[2] = (bf16)v.z; p[3] = (bf16)v.w;
    }
  }
  for (int i = t; i < 16 * 128; i += 256) {  // router^T, zero-padded cols 9..15
    int n = i >> 7, k = i & 127;
    rbt[i] = (n < NE) ? (bf16)router_w[k * NE + n] : (bf16)0.0f;
  }
  if (t < 18) auxl[t] = 0.f;
  __syncthreads();

  // x A-fragments live in registers for the whole kernel
  v16bf xa[4];
#pragma unroll
  for (int kc = 0; kc < 4; ++kc) xa[kc] = frag_a(xh + w * 16 * 128, lane, kc);

  { // logits = x@router_wx + cbias  (one WMMA column tile)
    float cb = (lm < NE) ? ws[WS_CB + lm] : 0.f;
    v8f lg = {cb, cb, cb, cb, cb, cb, cb, cb};
#pragma unroll
    for (int kc = 0; kc < 4; ++kc) lg = WMMA_BF16(xa[kc], frag_b(rbt, lane, 0, kc), lg);
    if (lm < NE) {
#pragma unroll
      for (int r = 0; r < 8; ++r) gat[w][hi * 8 + r][lm] = lg[r];
    }
  }
  __syncthreads();

  if (lane < 16) {  // per-token top-3 + softmax, aux stats
    float* gp = &gat[w][lane][0];
    float l[NE];
#pragma unroll
    for (int e = 0; e < NE; ++e) l[e] = gp[e];
    int i0 = 0;
#pragma unroll
    for (int e = 1; e < NE; ++e) if (l[e] > l[i0]) i0 = e;
    float v0 = l[i0]; l[i0] = -INFINITY;
    int i1 = 0;
#pragma unroll
    for (int e = 1; e < NE; ++e) if (l[e] > l[i1]) i1 = e;
    float v1 = l[i1]; l[i1] = -INFINITY;
    int i2 = 0;
#pragma unroll
    for (int e = 1; e < NE; ++e) if (l[e] > l[i2]) i2 = e;
    float v2 = l[i2];
    float e1 = __expf(v1 - v0), e2 = __expf(v2 - v0);
    float inv = 1.f / (1.f + e1 + e2);
    float g0 = inv, g1 = e1 * inv, g2 = e2 * inv;
#pragma unroll
    for (int e = 0; e < NE; ++e) gp[e] = 0.f;
    gp[i0] = g0; gp[i1] = g1; gp[i2] = g2;
    atomicAdd(&auxl[i0], 1.f);  atomicAdd(&auxl[i1], 1.f);  atomicAdd(&auxl[i2], 1.f);
    atomicAdd(&auxl[9 + i0], g0); atomicAdd(&auxl[9 + i1], g1); atomicAdd(&auxl[9 + i2], g2);
  }
  __syncthreads();
  if (t < 18) ws[WS_AUXP + blk * 18 + t] = auxl[t];   // deterministic partials

  v8f oacc[8];
#pragma unroll
  for (int d0 = 0; d0 < 8; ++d0) { v8f z = {0,0,0,0,0,0,0,0}; oacc[d0] = z; }

  bf16* hl = xh + w * 16 * 128;   // reuse x staging as per-wave h buffer

  for (int e = 0; e < NE; ++e) {
    const float* W1e = W1 + (size_t)e * 160 * 128;
    const float* W2e = W2 + (size_t)e * 128 * 128;
#pragma unroll
    for (int j = 0; j < 16; ++j) {  // coalesced load, transposed bf16 store to LDS
      int idx = (t + j * 256) * 4;
      int k = idx >> 7, h = idx & 127;
      float4 a = *(const float4*)(W1e + idx);
      w1t[(h + 0) * 128 + k] = (bf16)a.x;
      w1t[(h + 1) * 128 + k] = (bf16)a.y;
      w1t[(h + 2) * 128 + k] = (bf16)a.z;
      w1t[(h + 3) * 128 + k] = (bf16)a.w;
      float4 b = *(const float4*)(W2e + idx);
      w2t[(h + 0) * 128 + k] = (bf16)b.x;
      w2t[(h + 1) * 128 + k] = (bf16)b.y;
      w2t[(h + 2) * 128 + k] = (bf16)b.z;
      w2t[(h + 3) * 128 + k] = (bf16)b.w;
    }
    if (t < 128) { sbl[t] = ws[WS_SB + e * 128 + t]; b2l[t] = b2[e * 128 + t]; }
    __syncthreads();

    float gm[8];
#pragma unroll
    for (int r = 0; r < 8; ++r) gm[r] = gat[w][hi * 8 + r][e];

    // layer 1: h = relu(x @ W1x + style_bias)
#pragma unroll
    for (int n0 = 0; n0 < 8; ++n0) {
      float bv = sbl[n0 * 16 + lm];
      v8f c = {bv, bv, bv, bv, bv, bv, bv, bv};
#pragma unroll
      for (int kc = 0; kc < 4; ++kc) c = WMMA_BF16(xa[kc], frag_b(w1t, lane, n0, kc), c);
      int col = n0 * 16 + lm;
#pragma unroll
      for (int r = 0; r < 8; ++r) {
        float v = c[r] > 0.f ? c[r] : 0.f;
        hl[(hi * 8 + r) * 128 + col] = (bf16)v;
      }
    }
    __syncthreads();

    // layer 2: out_e = h @ W2 + b2, gated accumulate in f32
    v16bf ha[4];
#pragma unroll
    for (int kc = 0; kc < 4; ++kc) ha[kc] = frag_a(hl, lane, kc);
#pragma unroll
    for (int d0 = 0; d0 < 8; ++d0) {
      float bv = b2l[d0 * 16 + lm];
      v8f c = {bv, bv, bv, bv, bv, bv, bv, bv};
#pragma unroll
      for (int kc = 0; kc < 4; ++kc) c = WMMA_BF16(ha[kc], frag_b(w2t, lane, d0, kc), c);
#pragma unroll
      for (int r = 0; r < 8; ++r) oacc[d0][r] += gm[r] * c[r];
    }
    __syncthreads();
  }

  float* orow = out + ((size_t)blk * 128 + w * 16 + hi * 8) * 128;
#pragma unroll
  for (int d0 = 0; d0 < 8; ++d0)
#pragma unroll
    for (int r = 0; r < 8; ++r)
      orow[(size_t)r * 128 + d0 * 16 + lm] = oacc[d0][r];
}

// ---------------- K4: aux loss finalize (serial, deterministic) ----------------
__global__ void k_aux(const float* __restrict__ ws, float* __restrict__ out_aux) {
  if (threadIdx.x == 0 && blockIdx.x == 0) {
    float cnt[NE], gs[NE];
    for (int e = 0; e < NE; ++e) { cnt[e] = 0.f; gs[e] = 0.f; }
    for (int b = 0; b < 512; ++b)
      for (int e = 0; e < NE; ++e) {
        cnt[e] += ws[WS_AUXP + b * 18 + e];
        gs[e]  += ws[WS_AUXP + b * 18 + 9 + e];
      }
    const float T = 65536.f;
    float s = 0.f;
    for (int e = 0; e < NE; ++e)
      s += (cnt[e] * (9.f / (3.f * T))) * (gs[e] / T);
    out_aux[0] = 1.0e-4f * s;
  }
}

extern "C" void kernel_launch(void* const* d_in, const int* in_sizes, int n_in,
                              void* d_out, int out_size, void* d_ws, size_t ws_size,
                              hipStream_t stream) {
  const float* x        = (const float*)d_in[0];
  const float* Wr_w     = (const float*)d_in[1];
  const float* Wr_b     = (const float*)d_in[2];
  const float* Wl_w     = (const float*)d_in[3];
  const float* Wl_b     = (const float*)d_in[4];
  const float* router_w = (const float*)d_in[5];
  const float* router_b = (const float*)d_in[6];
  const float* styles   = (const float*)d_in[7];
  const float* W1       = (const float*)d_in[8];
  const float* b1       = (const float*)d_in[9];
  const float* W2       = (const float*)d_in[10];
  const float* b2       = (const float*)d_in[11];
  float* out = (float*)d_out;
  float* ws  = (float*)d_ws;

  k_market<<<NTOK / 256, 256, 0, stream>>>(x, Wr_w, Wr_b, Wl_w, ws);
  k_setup<<<1, 256, 0, stream>>>(Wl_b, router_w, router_b, styles, W1, b1, ws);
  k_moe<<<NTOK / 128, 256, 0, stream>>>(x, router_w, W1, W2, b2, ws, out);
  k_aux<<<1, 32, 0, stream>>>(ws, out + (size_t)NTOK * 128);
}
